// Deformable_Conv_46067819217314
// MI455X (gfx1250) — matrix operand order repaired and checked by reading, not compiled, not run on hardware
//
#include <hip/hip_runtime.h>

// ---------- types ----------
typedef __attribute__((ext_vector_type(16))) __bf16 v16bf;
typedef __attribute__((ext_vector_type(8)))  __bf16 v8bf;
typedef __attribute__((ext_vector_type(8)))  float  v8f;

// ---------- problem constants ----------
constexpr int Bn   = 4;
constexpr int Cc   = 64;        // input channels
constexpr int Hh   = 128;
constexpr int Ww   = 128;
constexpr int Oo   = 64;        // output channels
constexpr int NOFF = 36;        // DG*2*KK offset channels
constexpr int NOFFP = 48;       // padded to 3 N-tiles of 16
constexpr int Ktot = Cc * 9;    // 576 = GEMM K for both stages
constexpr int KSTEPS = Ktot / 32;   // 18
constexpr int PIX  = 32;        // pixels per workgroup (2 M-tiles)
constexpr int PST  = 592;       // panel row stride in bf16 (rows 16B-aligned)
constexpr int NPOS = PIX * 2 * 9;   // 576 (pixel, group, tap) descriptors

// ---------- fragment loaders ----------
// A-fragment (16-bit A 16x32 layout): lane 0..15 -> row M, K runs koff..koff+7
// and 16+koff..16+koff+7 (koff = 0); lanes 16..31 same rows with koff = 8.
__device__ __forceinline__ v16bf load_a_frag(const __bf16* p) {
    v8bf lo = *reinterpret_cast<const v8bf*>(p);
    v8bf hi = *reinterpret_cast<const v8bf*>(p + 16);
    return __builtin_shufflevector(lo, hi, 0,1,2,3,4,5,6,7,8,9,10,11,12,13,14,15);
}

// B-fragment from pre-converted bf16 weights (row-major [N][Ktot]):
// lane 0..15 -> col N, K = kbase..kbase+7 and kbase+16..kbase+23; lanes 16..31: K += 8.
__device__ __forceinline__ v16bf load_b_frag(const __bf16* __restrict__ p) {
    v8bf lo = *reinterpret_cast<const v8bf*>(p);
    v8bf hi = *reinterpret_cast<const v8bf*>(p + 16);
    return __builtin_shufflevector(lo, hi, 0,1,2,3,4,5,6,7,8,9,10,11,12,13,14,15);
}

// ---------- kernel 0: one-shot weight conversion into workspace ----------
// ws layout: [0, 48*576)  = w_off bf16, rows 36..47 zero
//            [48*576, +64*576) = w_def bf16
__global__ __launch_bounds__(256) void prep_weights_kernel(
    const float* __restrict__ w_off, const float* __restrict__ w_def,
    __bf16* __restrict__ ws)
{
    const int i = blockIdx.x * 256 + threadIdx.x;   // 0 .. 64*576-1
    if (i < NOFFP * Ktot) {
        const int n = i / Ktot;
        ws[i] = (__bf16)(n < NOFF ? w_off[i] : 0.0f);
    }
    if (i < Oo * Ktot) {
        ws[NOFFP * Ktot + i] = (__bf16)w_def[i];
    }
}

// ---------- kernel 1: fused deformable conv ----------
__global__ __launch_bounds__(256) void dcn_fused_kernel(
    const float*  __restrict__ x,       // [B,C,H,W]
    const __bf16* __restrict__ woffb,   // [48][576] bf16 (padded)
    const __bf16* __restrict__ wdefb,   // [64][576] bf16
    float* __restrict__ out)            // [B,O,H,W]
{
    __shared__ __align__(16) __bf16 panel[PIX * PST];   // 37888 B, reused twice
    __shared__ float offs[PIX * NOFF];                  // 4608 B
    __shared__ int   sy0[NPOS], sy1[NPOS], sx0[NPOS], sx1[NPOS];   // 9216 B
    __shared__ float sw00[NPOS], sw01[NPOS], sw10[NPOS], sw11[NPOS]; // 9216 B

    const int tid  = threadIdx.x;
    const int lane = tid & 31;
    const int wave = tid >> 5;

    const int tile = blockIdx.x;          // 2048 tiles = B * H * (W/32)
    const int w0   = (tile & 3) << 5;
    const int h    = (tile >> 2) & 127;
    const int b    = tile >> 9;

    const float* __restrict__ xb = x + (size_t)b * Cc * Hh * Ww;

    // ---------------- stage 1: im2col panel (plain 3x3 conv taps) ---------------
    for (int e = tid; e < PIX * Ktot; e += 256) {
        const int m  = e & 31;
        const int k  = e >> 5;
        const int c  = k / 9;
        const int kk = k - c * 9;
        const int yy = h + (kk / 3) - 1;
        const int xx = w0 + m + (kk % 3) - 1;
        float v = 0.0f;
        if ((unsigned)yy < (unsigned)Hh && (unsigned)xx < (unsigned)Ww)
            v = xb[(size_t)c * Hh * Ww + yy * Ww + xx];
        panel[m * PST + k] = (__bf16)v;
    }
    __syncthreads();

    const int koffA = (lane >> 4) * 8;    // A-fragment K sub-offset
    const int koffB = (lane >> 4) * 8;    // B-fragment K sub-offset
    const int nloc  = lane & 15;          // B/D column within tile
    const int mh    = (lane >> 4) * 8;    // D-fragment row base

    // ---------------- stage 2: offset conv (2 M-tiles x 3 N-tiles, waves 0..5) --
    if (wave < 6) {
        const int mt   = wave & 1;
        const int nt   = wave >> 1;                  // 0..2
        const int chan = nt * 16 + nloc;             // 0..47 (padded rows are zero)
        const int mrow = mt * 16 + (lane & 15);
        const __bf16* wrow = woffb + (size_t)chan * Ktot;
        v8f acc = {0.f, 0.f, 0.f, 0.f, 0.f, 0.f, 0.f, 0.f};
        for (int kb = 0; kb < KSTEPS; ++kb) {
            v16bf a  = load_a_frag(&panel[mrow * PST + kb * 32 + koffA]);
            v16bf bb = load_b_frag(wrow + kb * 32 + koffB);
            acc = __builtin_amdgcn_wmma_f32_16x16x32_bf16(
                false, a, false, bb, (short)0, acc, false, false);
        }
        if (chan < NOFF) {
#pragma unroll
            for (int r = 0; r < 8; ++r)
                offs[(mt * 16 + mh + r) * NOFF + chan] = acc[r];
        }
    }
    __syncthreads();

    // ---------------- stage 3: bilinear descriptors per (pixel, group, tap) -----
    for (int p = tid; p < NPOS; p += 256) {
        const int m   = p / 18;
        const int rem = p - m * 18;
        const int g   = rem / 9;
        const int kk  = rem - g * 9;
        const float oy = offs[m * NOFF + g * 18 + kk * 2 + 0];
        const float ox = offs[m * NOFF + g * 18 + kk * 2 + 1];
        const float py = (float)h        + oy + (float)(kk / 3 - 1);
        const float px = (float)(w0 + m) + ox + (float)(kk % 3 - 1);
        const float fy = floorf(py);
        const float fx = floorf(px);
        const int   y0 = (int)fy, x0 = (int)fx;
        const int   y1 = y0 + 1,  x1 = x0 + 1;
        const float dy = py - fy, dx = px - fx;
        const float vy0 = ((unsigned)y0 < (unsigned)Hh) ? 1.f : 0.f;
        const float vy1 = ((unsigned)y1 < (unsigned)Hh) ? 1.f : 0.f;
        const float vx0 = ((unsigned)x0 < (unsigned)Ww) ? 1.f : 0.f;
        const float vx1 = ((unsigned)x1 < (unsigned)Ww) ? 1.f : 0.f;
        sw00[p] = (1.f - dy) * (1.f - dx) * vy0 * vx0;
        sw01[p] = (1.f - dy) * dx         * vy0 * vx1;
        sw10[p] = dy * (1.f - dx)         * vy1 * vx0;
        sw11[p] = dy * dx                 * vy1 * vx1;
        sy0[p] = y0 < 0 ? 0 : (y0 > Hh - 1 ? Hh - 1 : y0);
        sy1[p] = y1 < 0 ? 0 : (y1 > Hh - 1 ? Hh - 1 : y1);
        sx0[p] = x0 < 0 ? 0 : (x0 > Ww - 1 ? Ww - 1 : x0);
        sx1[p] = x1 < 0 ? 0 : (x1 > Ww - 1 ? Ww - 1 : x1);
    }
    __syncthreads();

    // ---------------- stage 4: deformable bilinear gather into panel -------------
    for (int e = tid; e < PIX * Ktot; e += 256) {
        const int m  = e & 31;
        const int k  = e >> 5;
        const int c  = k / 9;             // c = g*32 + cg
        const int kk = k - c * 9;
        const int g  = c >> 5;
        const int p  = m * 18 + g * 9 + kk;
        const float* __restrict__ ch = xb + (size_t)c * Hh * Ww;
        const int y0 = sy0[p], y1 = sy1[p], x0 = sx0[p], x1 = sx1[p];
        const float v = sw00[p] * ch[y0 * Ww + x0]
                      + sw01[p] * ch[y0 * Ww + x1]
                      + sw10[p] * ch[y1 * Ww + x0]
                      + sw11[p] * ch[y1 * Ww + x1];
        panel[m * PST + k] = (__bf16)v;
    }
    __syncthreads();

    // ---------------- stage 5: main GEMM (2 M-tiles x 4 N-tiles, all 8 waves) ---
    {
        const int mt   = wave & 1;
        const int nt   = wave >> 1;                  // 0..3
        const int chan = nt * 16 + nloc;             // 0..63
        const int mrow = mt * 16 + (lane & 15);
        const __bf16* wrow = wdefb + (size_t)chan * Ktot;
        v8f acc = {0.f, 0.f, 0.f, 0.f, 0.f, 0.f, 0.f, 0.f};
        for (int kb = 0; kb < KSTEPS; ++kb) {
            v16bf a  = load_a_frag(&panel[mrow * PST + kb * 32 + koffA]);
            v16bf bb = load_b_frag(wrow + kb * 32 + koffB);
            acc = __builtin_amdgcn_wmma_f32_16x16x32_bf16(
                false, a, false, bb, (short)0, acc, false, false);
        }
        float* __restrict__ orow =
            out + (((size_t)b * Oo + chan) * Hh + h) * Ww + w0 + mt * 16;
#pragma unroll
        for (int r = 0; r < 8; ++r)
            orow[mh + r] = acc[r];
    }
}

extern "C" void kernel_launch(void* const* d_in, const int* in_sizes, int n_in,
                              void* d_out, int out_size, void* d_ws, size_t ws_size,
                              hipStream_t stream) {
    (void)in_sizes; (void)n_in; (void)ws_size; (void)out_size;
    const float* x     = (const float*)d_in[0];
    const float* w_off = (const float*)d_in[1];
    const float* w_def = (const float*)d_in[2];
    float* out = (float*)d_out;
    __bf16* wsb = (__bf16*)d_ws;

    // kernel 0: weights -> bf16 (w_off padded to 48 rows)
    hipLaunchKernelGGL(prep_weights_kernel, dim3((Oo * Ktot + 255) / 256), dim3(256),
                       0, stream, w_off, w_def, wsb);

    // kernel 1: fused deformable conv
    const int tiles = Bn * Hh * (Ww / PIX);   // 2048
    hipLaunchKernelGGL(dcn_fused_kernel, dim3(tiles), dim3(256), 0, stream,
                       x, wsb, wsb + NOFFP * Ktot, out);
}
